// UnifiedSequentialTokenizer_15410342658371
// MI455X (gfx1250) — compile-verified, matrix operand order checked by reading, not run on hardware
//
#include <hip/hip_runtime.h>
#include <hip/hip_bf16.h>
#include <cstdint>

typedef __attribute__((ext_vector_type(16))) __bf16 v16bf;
typedef __attribute__((ext_vector_type(8)))  __bf16 v8bf;
typedef __attribute__((ext_vector_type(8)))  float  v8f;

#define B_    32
#define L_    512
#define H_    256
#define SIXH  1536
#define FOURH 1024
#define M_    1024

// ---------------------------------------------------------------------------
// Weight conversion: f32 row-major -> bf16 column-major (N-major over K).
// ---------------------------------------------------------------------------
__global__ void convert_w1(const float* __restrict__ W1, __bf16* __restrict__ W1t) {
  int tid = blockIdx.x * 256 + threadIdx.x;          // FOURH*SIXH total
  if (tid >= FOURH * SIXH) return;
  int n = tid / SIXH, k = tid % SIXH;
  W1t[tid] = (__bf16)W1[k * FOURH + n];
}

__global__ void convert_w2(const float* __restrict__ W2, __bf16* __restrict__ W2t) {
  int tid = blockIdx.x * 256 + threadIdx.x;          // H*FOURH total
  if (tid >= H_ * FOURH) return;
  int n = tid / FOURH, k = tid % FOURH;
  W2t[tid] = (__bf16)W2[k * H_ + n];
}

// ---------------------------------------------------------------------------
// Stage 1: gather 6 embedding segments, LayerNorm over 6H, emit bf16 Xn row.
// ---------------------------------------------------------------------------
__global__ void assemble_ln(const int* __restrict__ t0, const int* __restrict__ t1,
                            const int* __restrict__ t2, const int* __restrict__ t3,
                            const int* __restrict__ tg, const int* __restrict__ gid,
                            const float* __restrict__ tok, const float* __restrict__ tim,
                            const float* __restrict__ grp,
                            const float* __restrict__ lng, const float* __restrict__ lnb,
                            __bf16* __restrict__ Xn) {
  __shared__ float ssum[256], ssq[256];
  const int r = blockIdx.x;                 // row in [0, B*L)
  const int i = threadIdx.x;                // column within H
  int a0 = t0[r], a1 = t1[r], a2 = t2[r], a3 = t3[r];
  int tgi = tg[r]; tgi = tgi < 0 ? 0 : (tgi > 64 ? 64 : tgi);
  int gi  = gid[r];
  float v[6];
  v[0] = tok[(size_t)a0 * H_ + i];
  v[1] = tok[(size_t)a1 * H_ + i];
  v[2] = tok[(size_t)a2 * H_ + i];
  v[3] = tok[(size_t)a3 * H_ + i];
  v[4] = tim[(size_t)tgi * H_ + i];
  v[5] = grp[(size_t)gi  * H_ + i];
  float s = 0.f, q = 0.f;
#pragma unroll
  for (int j = 0; j < 6; ++j) { s += v[j]; q += v[j] * v[j]; }
  ssum[i] = s; ssq[i] = q;
  __syncthreads();
  for (int off = 128; off > 0; off >>= 1) {
    if (i < off) { ssum[i] += ssum[i + off]; ssq[i] += ssq[i + off]; }
    __syncthreads();
  }
  const float mu  = ssum[0] * (1.0f / SIXH);
  const float var = ssq[0] * (1.0f / SIXH) - mu * mu;
  const float rs  = rsqrtf(var + 1e-5f);
#pragma unroll
  for (int j = 0; j < 6; ++j) {
    int c = j * 256 + i;
    Xn[(size_t)r * SIXH + c] = (__bf16)((v[j] - mu) * rs * lng[c] + lnb[c]);
  }
}

// ---------------------------------------------------------------------------
// bf16 WMMA GEMM with async-to-LDS double-buffered B tiles.
//   Block = 8 waves -> 128 rows x 64 cols.  Per 32-deep k-chunk the block's
//   B tile (64 cols x 32 k bf16, column stride padded 64B->80B against bank
//   conflicts) is fetched with one global_load_async_to_lds_b128 per wave
//   (ASYNCcnt), overlapped with compute on the other buffer.
//   A fragment (16x32): lane&15 = M row; half-wave = K {0..7,16..23} vs
//   {8..15,24..31} -> two contiguous 16B global loads.  B fragments come from
//   LDS via ds_load_b128.  C/D: elem r -> M=r+8*half, N=lane&15.
// ---------------------------------------------------------------------------
#define COLPAD 80   // bytes per (column, k-chunk) strip: 64 data + 16 pad

template <int KDIM, int NDIM, bool SILU, bool OUTBF16>
__global__ void wmma_gemm(const __bf16* __restrict__ A, const __bf16* __restrict__ Bm,
                          const float* __restrict__ bias,
                          __bf16* __restrict__ Obf, float* __restrict__ Of) {
  constexpr int NK = KDIM / 32;
  __shared__ __align__(16) unsigned char sB[2][64 * COLPAD];

  const int lane = threadIdx.x & 31;
  const int wave = threadIdx.x >> 5;
  const int nBlocksN = NDIM / 64;
  const int bm = blockIdx.x / nBlocksN;
  const int bn = blockIdx.x % nBlocksN;
  const int half = lane >> 4;
  const int idx  = lane & 15;

  // async-copy mapping: thread t moves 16B piece (t&3) of column (t>>2)
  const int cCol   = threadIdx.x >> 2;
  const int cPiece = threadIdx.x & 3;
  const __bf16* gB = Bm + (size_t)(bn * 64 + cCol) * KDIM + cPiece * 8;
  const unsigned ldsDst0 = (unsigned)(uintptr_t)&sB[0][cCol * COLPAD + cPiece * 16];
  const unsigned ldsDst1 = (unsigned)(uintptr_t)&sB[1][cCol * COLPAD + cPiece * 16];

  const __bf16* aRow = A + (size_t)(bm * 128 + wave * 16 + idx) * KDIM + half * 8;

  v8f acc[4] = {};

  {  // preload chunk 0 into buffer 0
    unsigned long long src = (unsigned long long)gB;
    asm volatile("global_load_async_to_lds_b128 %0, %1, off"
                 :: "v"(ldsDst0), "v"(src) : "memory");
  }

  for (int kc = 0; kc < NK; ++kc) {
    if (kc + 1 < NK) {
      unsigned dst = ((kc + 1) & 1) ? ldsDst1 : ldsDst0;
      unsigned long long src = (unsigned long long)(gB + (kc + 1) * 32);
      asm volatile("global_load_async_to_lds_b128 %0, %1, off"
                   :: "v"(dst), "v"(src) : "memory");
      asm volatile("s_wait_asynccnt 0x1" ::: "memory");   // my chunk-kc slice done
    } else {
      asm volatile("s_wait_asynccnt 0x0" ::: "memory");
    }
    __syncthreads();                     // every wave's slice of chunk kc landed

    const int k0 = kc * 32;
    v8bf alo = *(const v8bf*)(aRow + k0);
    v8bf ahi = *(const v8bf*)(aRow + k0 + 16);
    if (k0 + 64 < KDIM) __builtin_prefetch(aRow + k0 + 64, 0, 3);
    v16bf af = __builtin_shufflevector(alo, ahi, 0, 1, 2, 3, 4, 5, 6, 7,
                                       8, 9, 10, 11, 12, 13, 14, 15);
    const unsigned char* base = &sB[kc & 1][0];
    v8bf bl[4], bh[4];
#pragma unroll
    for (int s = 0; s < 4; ++s) {        // all LDS reads up front
      const unsigned char* cptr = base + (s * 16 + idx) * COLPAD + half * 16;
      bl[s] = *(const v8bf*)(cptr);
      bh[s] = *(const v8bf*)(cptr + 32);
    }
#pragma unroll
    for (int s = 0; s < 4; ++s) {
      v16bf bf = __builtin_shufflevector(bl[s], bh[s], 0, 1, 2, 3, 4, 5, 6, 7,
                                         8, 9, 10, 11, 12, 13, 14, 15);
      acc[s] = __builtin_amdgcn_wmma_f32_16x16x32_bf16(
          /*neg_a=*/false, af, /*neg_b=*/false, bf,
          /*c_mod=*/(short)0, acc[s], /*reuse_a=*/false, /*reuse_b=*/false);
    }
    __syncthreads();                     // all reads of this buffer done before
  }                                      // it is overwritten two chunks later

#pragma unroll
  for (int s = 0; s < 4; ++s) {
    const int n  = bn * 64 + s * 16 + idx;
    const float bv = bias[n];
#pragma unroll
    for (int rr = 0; rr < 8; ++rr) {
      const int m = bm * 128 + wave * 16 + rr + half * 8;
      float val = acc[s][rr] + bv;
      if (SILU) val = val * (1.0f / (1.0f + __expf(-val)));
      if (OUTBF16) Obf[(size_t)m * NDIM + n] = (__bf16)val;
      else         Of [(size_t)m * NDIM + n] = val;
    }
  }
}

// ---------------------------------------------------------------------------
// Stage 3: merge_with_sep. One block per batch.
// ---------------------------------------------------------------------------
__global__ void merge_kernel(const int* __restrict__ gid, const int* __restrict__ mask,
                             const float* __restrict__ ev, const float* __restrict__ pos,
                             const float* __restrict__ sep,
                             float* __restrict__ merged, float* __restrict__ mout) {
  __shared__ int s_nxt[L_];
  __shared__ int s_tgt[2 * L_];
  __shared__ int s_cnt;
  const int b = blockIdx.x;
  const int* gm = gid + b * L_;
  const int* mm = mask + b * L_;

  if (threadIdx.x == 0) {
    int carry = -1;
    for (int l = L_ - 1; l >= 0; --l) { s_nxt[l] = carry; if (mm[l]) carry = gm[l]; }
    int cnt = 0;
    for (int p = 0; p < 2 * L_; ++p) {
      const int l  = p >> 1;
      const int mv = mm[l] != 0;
      const int mc = (p & 1) ? (mv && s_nxt[l] >= 0 && s_nxt[l] != gm[l]) : mv;
      s_tgt[p] = mc ? cnt++ : -1;
    }
    s_cnt = cnt;
  }
  __syncthreads();

  float* mg = merged + (size_t)b * M_ * H_;
  for (int j = threadIdx.x; j < M_ * H_; j += 256) mg[j] = 0.0f;
  for (int j = threadIdx.x; j < M_;      j += 256) mout[b * M_ + j] = 0.0f;
  __syncthreads();

  const int lane = threadIdx.x & 31;
  const int wave = threadIdx.x >> 5;
  const int cnt  = s_cnt;
  for (int p = wave; p < 2 * L_; p += 8) {
    const int rk = s_tgt[p];
    if (rk < 0) continue;
    const int t = M_ - cnt + rk;
    if (t < 0 || t >= M_) continue;
    const int l = p >> 1;
    const float* src = (p & 1) ? sep : (ev + (size_t)(b * L_ + l) * H_);
    for (int h = lane; h < H_; h += 32)
      mg[t * H_ + h] = src[h] + pos[t * H_ + h];
    if (lane == 0) mout[b * M_ + t] = 1.0f;
  }
}

// ---------------------------------------------------------------------------
extern "C" void kernel_launch(void* const* d_in, const int* in_sizes, int n_in,
                              void* d_out, int out_size, void* d_ws, size_t ws_size,
                              hipStream_t stream) {
  (void)in_sizes; (void)n_in; (void)out_size; (void)ws_size;
  const int*   tok0 = (const int*)d_in[0];
  const int*   tok1 = (const int*)d_in[1];
  const int*   tok2 = (const int*)d_in[2];
  const int*   tok3 = (const int*)d_in[3];
  const int*   tgap = (const int*)d_in[4];
  const int*   gid  = (const int*)d_in[5];
  const int*   msk  = (const int*)d_in[6];
  const float* tokE = (const float*)d_in[7];
  const float* timE = (const float*)d_in[8];
  const float* grpE = (const float*)d_in[9];
  const float* posE = (const float*)d_in[10];
  const float* sepT = (const float*)d_in[11];
  const float* lnG  = (const float*)d_in[12];
  const float* lnB  = (const float*)d_in[13];
  const float* W1   = (const float*)d_in[14];
  const float* b1   = (const float*)d_in[15];
  const float* W2   = (const float*)d_in[16];
  const float* b2   = (const float*)d_in[17];

  // Workspace layout (256B-aligned offsets; ev aliases Xn, which is dead then)
  char* ws = (char*)d_ws;
  __bf16* Xn  = (__bf16*)(ws);                                    // 50,331,648 B
  float*  ev  = (float*)(ws);                                     // alias, 16.8 MB
  __bf16* W1t = (__bf16*)(ws + 50331648);                         //  3,145,728 B
  __bf16* W2t = (__bf16*)(ws + 50331648 + 3145728);               //    524,288 B
  __bf16* Y   = (__bf16*)(ws + 50331648 + 3145728 + 524288);      // 33,554,432 B

  convert_w1<<<(FOURH * SIXH + 255) / 256, 256, 0, stream>>>(W1, W1t);
  convert_w2<<<(H_ * FOURH + 255) / 256, 256, 0, stream>>>(W2, W2t);

  assemble_ln<<<B_ * L_, 256, 0, stream>>>(tok0, tok1, tok2, tok3, tgap, gid,
                                           tokE, timE, grpE, lnG, lnB, Xn);

  // GEMM1 + SiLU: blocks = (16384/128) * (1024/64) = 2048
  wmma_gemm<SIXH, FOURH, true, true>
      <<<2048, 256, 0, stream>>>(Xn, W1t, b1, Y, nullptr);

  // GEMM2: blocks = (16384/128) * (256/64) = 512
  wmma_gemm<FOURH, H_, false, false>
      <<<512, 256, 0, stream>>>(Y, W2t, b2, nullptr, ev);

  float* merged = (float*)d_out;
  float* moutp  = merged + (size_t)B_ * M_ * H_;
  merge_kernel<<<B_, 256, 0, stream>>>(gid, msk, ev, posE, sepT, merged, moutp);
}
